// Qwen2GLADecoderLayer_62130996904030
// MI455X (gfx1250) — compile-verified, matrix operand order checked by reading, not run on hardware
//
#include <hip/hip_runtime.h>
#include <hip/hip_bf16.h>
#include <math.h>

// ---------------------------------------------------------------------------
// Types for CDNA5 WMMA / TDM
// ---------------------------------------------------------------------------
typedef __attribute__((ext_vector_type(16))) _Float16 v16h;
typedef __attribute__((ext_vector_type(8)))  _Float16 v8h;
typedef __attribute__((ext_vector_type(8)))  float    v8f;
typedef __attribute__((ext_vector_type(4)))  unsigned u32x4;
typedef __attribute__((ext_vector_type(8)))  int      i32x8;
typedef __attribute__((ext_vector_type(4)))  int      i32x4;

#define DM      2048
#define IM      5504
#define BB      2
#define TSEQ    4096
#define NTOK    (BB * TSEQ)      // 8192
#define HEADS   8
#define HD      256              // head dim (HK == HV)
#define CHUNK   64
#define VB      32               // V-block per GLA workgroup

// ---------------------------------------------------------------------------
// Fragment helpers (16-bit A-layout: lane m = lane&15, K-halves split by lane>>4)
// ---------------------------------------------------------------------------
static __device__ inline v16h ld_frag(const _Float16* base, int stride, int lane, int k0) {
  const int m    = lane & 15;
  const int koff = ((lane >> 4) << 3) + k0;
  const _Float16* p = base + m * stride + koff;
  union { v16h v; v8h h[2]; } u;
  u.h[0] = *(const v8h*)(p);
  u.h[1] = *(const v8h*)(p + 16);
  return u.v;
}

static __device__ inline v8f wmma16(v16h a, v16h b, v8f c) {
  return __builtin_amdgcn_wmma_f32_16x16x32_f16(false, a, false, b, (short)0, c, false, false);
}

static __device__ inline v8f vzero8() { v8f z; for (int i = 0; i < 8; ++i) z[i] = 0.f; return z; }
static __device__ inline float siluf(float x) { return x / (1.f + expf(-x)); }

// LDS byte offset from a generic pointer: its low 32 bits are
// the wave-relative offset VDST / D#.lds_addr expect.
static __device__ inline unsigned lds_off32(const void* p) {
  return (unsigned)(unsigned long long)p;
}

// Per-lane async copy of 16 bytes global -> LDS (tracked with ASYNCcnt).
static __device__ inline void async_ld_b128(unsigned ldsoff, const void* gaddr) {
  asm volatile("global_load_async_to_lds_b128 %0, %1, off"
               :: "v"(ldsoff), "v"((unsigned long long)gaddr) : "memory");
}

// Tensor Data Mover: 2D tile (rows x elems of 2-byte data) global -> LDS with
// LDS padding of 4 DWORDs after every 32 DWORDs (row 128B -> stride 144B = 72 halves).
// This toolchain exposes the 6-arg builtin:
//   (u32x4 g0, i32x8 g1, i32x4 g2, i32x4 g3, i32x8 extra, i32 cpol)
static __device__ inline void tdm_load_tile_2d(const void* gsrc, unsigned ldsaddr,
                                               unsigned tensor_d0, unsigned tensor_d1,
                                               unsigned tile_d0, unsigned tile_d1,
                                               unsigned long long stride0_elems) {
  unsigned long long ga = (unsigned long long)gsrc;
  u32x4 g0;
  g0[0] = 1u;                                                    // count=1 (user D#)
  g0[1] = ldsaddr;                                               // lds_addr
  g0[2] = (unsigned)(ga & 0xFFFFFFFFu);                          // global_addr[31:0]
  g0[3] = (unsigned)((ga >> 32) & 0x01FFFFFFu) | (2u << 30);     // addr[56:32] | type=2
  i32x8 g1;
  g1[0] = (int)((1u << 16) | (1u << 20) | (4u << 22) | (3u << 25)); // 2B elems, pad_en,
                                                                    // interval=32dw, amount=4dw
  g1[1] = (int)((tensor_d0 & 0xFFFFu) << 16);                      // tensor_dim0[15:0]
  g1[2] = (int)(((tensor_d0 >> 16) & 0xFFFFu) |
                ((tensor_d1 & 0xFFFFu) << 16));                    // dim0[31:16] | dim1[15:0]
  g1[3] = (int)(((tensor_d1 >> 16) & 0xFFFFu) | (tile_d0 << 16));  // dim1[31:16] | tile_dim0
  g1[4] = (int)(tile_d1 & 0xFFFFu);                                // tile_dim1 (tile_dim2=0)
  g1[5] = (int)(unsigned)(stride0_elems & 0xFFFFFFFFu);            // dim0_stride[31:0]
  g1[6] = (int)(unsigned)((stride0_elems >> 32) & 0xFFFFu);        // dim0_stride[47:32]
  g1[7] = 0;
  i32x4 z4; z4[0] = z4[1] = z4[2] = z4[3] = 0;
  i32x8 z8; for (int i = 0; i < 8; ++i) z8[i] = 0;
  __builtin_amdgcn_tensor_load_to_lds(g0, g1, z4, z4, z8, 0);
}

// ---------------------------------------------------------------------------
// Generic NT GEMM: C[M,N] = A[M,K](f16,row) * BT[N,K](f16,row)  (+resid), f32 out
// Tile: 128(M) x 64(N) x 64(K); double-buffered pipeline:
//   A tile via per-lane GLOBAL_LOAD_ASYNC_TO_LDS_B128 (ASYNCcnt),
//   B tile via one TENSOR_LOAD_TO_LDS (TENSORcnt, wave 0).
// 8 waves, each wave 32x32 via 2x2 WMMA tiles, 8 WMMAs per K-tile.
// ---------------------------------------------------------------------------
#define GBM 128
#define GBN 64
#define GBK 64
#define GSTR 72                        // padded LDS row stride (halves)
#define ATILE_H (GBM * GSTR)           // 9216 halves
#define BTILE_H (GBN * GSTR)           // 4608 halves
#define BUF_H   (ATILE_H + BTILE_H)    // 13824 halves per buffer

template<int EPI>
__global__ __launch_bounds__(256)
void glaql_gemm(const _Float16* __restrict__ A, const _Float16* __restrict__ BT,
                float* __restrict__ C, const float* __restrict__ resid,
                int M, int N, int K) {
  __shared__ _Float16 sm[2 * BUF_H];   // ~54 KB
  const int tid  = threadIdx.x;
  const int lane = tid & 31;
  const int wave = tid >> 5;     // 0..7
  const int wm   = wave >> 1;    // 0..3 : 32-row strip
  const int wn   = wave & 1;     // 0..1 : 32-col strip
  const int m0   = blockIdx.y * GBM;
  const int n0   = blockIdx.x * GBN;

  v8f acc[2][2];
  for (int i = 0; i < 2; ++i) for (int j = 0; j < 2; ++j) acc[i][j] = vzero8();

  // A tile: 128 rows x 64 halves = 1024 x 16B chunks; 4 per thread (async)
  auto issueA = [&](int kt, int buf) {
    const int k0 = kt * GBK;
    _Float16* As = sm + buf * BUF_H;
    for (int c = 0; c < 4; ++c) {
      int id  = tid + c * 256;
      int row = id >> 3, seg = id & 7;
      async_ld_b128(lds_off32(As + row * GSTR + seg * 8),
                    A + (size_t)(m0 + row) * K + k0 + seg * 8);
    }
  };
  // B tile: one TDM op (issued from wave 0 only; TDM ignores EXEC)
  auto issueB = [&](int kt, int buf) {
    const int k0 = kt * GBK;
    _Float16* Bs = sm + buf * BUF_H + ATILE_H;
    tdm_load_tile_2d(BT + (size_t)n0 * K + k0, lds_off32(Bs),
                     (unsigned)(K - k0), (unsigned)(N - n0),
                     GBK, GBN, (unsigned long long)K);
  };

  const int KT = K / GBK;
  issueA(0, 0);
  if (wave == 0) issueB(0, 0);
  for (int kt = 0; kt < KT; ++kt) {
    const int cur = kt & 1;
    if (kt + 1 < KT) {
      issueA(kt + 1, cur ^ 1);
      if (wave == 0) issueB(kt + 1, cur ^ 1);
      asm volatile("s_wait_asynccnt 4" ::: "memory");          // prev A tile landed
      if (wave == 0) __builtin_amdgcn_s_wait_tensorcnt(1);     // prev B tile landed
    } else {
      asm volatile("s_wait_asynccnt 0" ::: "memory");
      if (wave == 0) __builtin_amdgcn_s_wait_tensorcnt(0);
    }
    __syncthreads();
    const _Float16* As = sm + cur * BUF_H;
    const _Float16* Bs = As + ATILE_H;
    for (int s = 0; s < 2; ++s) {
      v16h bf[2];
      for (int j = 0; j < 2; ++j)
        bf[j] = ld_frag(Bs + (wn * 32 + j * 16) * GSTR, GSTR, lane, s * 32);
      for (int i = 0; i < 2; ++i) {
        v16h af = ld_frag(As + (wm * 32 + i * 16) * GSTR, GSTR, lane, s * 32);
        for (int j = 0; j < 2; ++j)
          acc[i][j] = wmma16(af, bf[j], acc[i][j]);
      }
    }
    __syncthreads();
  }

  const int n  = lane & 15;
  const int mr = (lane >> 4) * 8;
  for (int i = 0; i < 2; ++i)
    for (int j = 0; j < 2; ++j) {
      int row = m0 + wm * 32 + i * 16 + mr;
      int col = n0 + wn * 32 + j * 16 + n;
      for (int r = 0; r < 8; ++r) {
        size_t idx = (size_t)(row + r) * N + col;
        float v = acc[i][j][r];
        if (EPI == 1) v += resid[idx];
        C[idx] = v;
      }
    }
}

// ---------------------------------------------------------------------------
// Weight convert + transpose: w[K,N] f32 -> wT[N,K] f16
// ---------------------------------------------------------------------------
__global__ void glaql_convT(const float* __restrict__ w, _Float16* __restrict__ wT,
                            int K, int N) {
  size_t total = (size_t)K * N;
  for (size_t i = (size_t)blockIdx.x * blockDim.x + threadIdx.x; i < total;
       i += (size_t)gridDim.x * blockDim.x) {
    size_t kx = i / N, nx = i % N;
    wT[nx * K + kx] = (_Float16)w[i];
  }
}

// ---------------------------------------------------------------------------
// RMSNorm over D=2048 -> f16 copy. One block (256 thr) per token, 8 ch/thread.
// ---------------------------------------------------------------------------
__global__ __launch_bounds__(256)
void glaql_rmsnorm16(const float* __restrict__ x, const float* __restrict__ w,
                     _Float16* __restrict__ out, float eps) {
  __shared__ float red[256];
  const size_t tok = blockIdx.x;
  const int tid = threadIdx.x;
  const float* xr = x + tok * DM;
  float vals[8], s = 0.f;
  for (int i = 0; i < 8; ++i) { float v = xr[tid + i * 256]; vals[i] = v; s += v * v; }
  red[tid] = s; __syncthreads();
  for (int st = 128; st > 0; st >>= 1) { if (tid < st) red[tid] += red[tid + st]; __syncthreads(); }
  float inv = rsqrtf(red[0] / (float)DM + eps);
  for (int i = 0; i < 8; ++i) {
    int d = tid + i * 256;
    out[tok * DM + d] = (_Float16)(vals[i] * inv * w[d]);
  }
}

// ---------------------------------------------------------------------------
// Short causal conv (W=4) + SiLU : lin f32 -> f16
// ---------------------------------------------------------------------------
__global__ void glaql_conv_silu(const float* __restrict__ lin, const float* __restrict__ cw,
                                _Float16* __restrict__ out16) {
  size_t total = (size_t)NTOK * DM;
  for (size_t i = (size_t)blockIdx.x * blockDim.x + threadIdx.x; i < total;
       i += (size_t)gridDim.x * blockDim.x) {
    size_t tok = i / DM; int d = (int)(i % DM);
    int t = (int)(tok % TSEQ);
    size_t bbase = (tok - (size_t)t) * DM;
    float y = 0.f;
    for (int j = 0; j < 4; ++j) {
      int ts = t + j - 3;
      if (ts >= 0) y += lin[bbase + (size_t)ts * DM + d] * cw[d * 4 + j];
    }
    out16[i] = (_Float16)siluf(y);
  }
}

// ---------------------------------------------------------------------------
// gk low-rank: stage 1  tmp[tok,16] = x16[tok,:] @ w1[D,16]
// ---------------------------------------------------------------------------
__global__ __launch_bounds__(256)
void glaql_gk1(const _Float16* __restrict__ x16, const float* __restrict__ w1,
               float* __restrict__ tmp) {
  __shared__ float red[256];
  const size_t tok = blockIdx.x;
  const int tid = threadIdx.x;
  float acc[16];
  for (int r = 0; r < 16; ++r) acc[r] = 0.f;
  for (int d = tid; d < DM; d += 256) {
    float xv = (float)x16[tok * DM + d];
    for (int r = 0; r < 16; ++r) acc[r] += xv * w1[(size_t)d * 16 + r];
  }
  for (int r = 0; r < 16; ++r) {
    red[tid] = acc[r]; __syncthreads();
    for (int st = 128; st > 0; st >>= 1) { if (tid < st) red[tid] += red[tid + st]; __syncthreads(); }
    if (tid == 0) tmp[tok * 16 + r] = red[0];
    __syncthreads();
  }
}

// stage 2: gk = log_sigmoid(tmp @ w2 + b) / 16
__global__ void glaql_gk2(const float* __restrict__ tmp, const float* __restrict__ w2,
                          const float* __restrict__ bias, float* __restrict__ gkout) {
  size_t total = (size_t)NTOK * DM;
  for (size_t i = (size_t)blockIdx.x * blockDim.x + threadIdx.x; i < total;
       i += (size_t)gridDim.x * blockDim.x) {
    size_t tok = i / DM; int d = (int)(i % DM);
    float s = bias[d];
    for (int r = 0; r < 16; ++r) s += tmp[tok * 16 + r] * w2[(size_t)r * DM + d];
    float ls = (s >= 0.f) ? -log1pf(expf(-s)) : (s - log1pf(expf(s)));
    gkout[i] = ls * 0.0625f;
  }
}

// ---------------------------------------------------------------------------
// Chunked GLA. One workgroup per (b*h, v-block of 32). State S^T[32][256] f32
// persistent in LDS; 64 sequential chunks; all 4 matmuls via WMMA.
// ---------------------------------------------------------------------------
__global__ __launch_bounds__(256)
void glaql_gla(const _Float16* __restrict__ q16, const _Float16* __restrict__ k16,
               const _Float16* __restrict__ v16, const float* __restrict__ gk,
               float* __restrict__ o) {
  extern __shared__ char smem_raw[];
  float*     S   = (float*)smem_raw;             // [VB][256]   S^T (v-major)
  float*     ebl = S + VB * 256;                 // [256] exp(b_last)
  _Float16*  S16 = (_Float16*)(ebl + 256);       // [VB][256]   f16 copy of S^T
  _Float16*  qs  = S16 + VB * 256;               // [64][256]   q * exp(b) * scale
  _Float16*  ki  = qs + 64 * 256;                // [64][256]   k * exp(-b)
  _Float16*  knT = ki + 64 * 256;                // [256][64]   (k*exp(bl-b))^T
  _Float16*  vT  = knT + 256 * 64;               // [VB][64]    v-block^T
  _Float16*  Am  = vT + VB * 64;                 // [64][72]    masked A, f16

  const int tid = threadIdx.x, lane = tid & 31, wave = tid >> 5;
  const int bh = blockIdx.x;
  const int h = bh % HEADS, b = bh / HEADS;
  const int vbase = blockIdx.y * VB;
  const float scale = 0.0625f;  // HD^-0.5 = 1/16

  for (int i = tid; i < VB * 256; i += 256) { S[i] = 0.f; S16[i] = (_Float16)0.f; }
  __syncthreads();

  const int NCK = TSEQ / CHUNK;
  for (int c = 0; c < NCK; ++c) {
    // ---- preprocess: thread tid owns k-channel kk ----
    const int kk = tid;
    const size_t base = ((size_t)(b * TSEQ + c * CHUNK)) * DM + (size_t)h * HD + kk;
    float cum = 0.f;
    for (int t = 0; t < CHUNK; ++t) {
      size_t idx = base + (size_t)t * DM;
      cum += gk[idx];
      qs[t * 256 + kk] = (_Float16)((float)q16[idx] * expf(cum) * scale);
      ki[t * 256 + kk] = (_Float16)((float)k16[idx] * expf(-cum));
    }
    const float blast = cum;
    ebl[kk] = expf(blast);
    float cum2 = 0.f;
    for (int t = 0; t < CHUNK; ++t) {
      size_t idx = base + (size_t)t * DM;
      cum2 += gk[idx];
      knT[kk * 64 + t] = (_Float16)((float)k16[idx] * expf(blast - cum2));
    }
    for (int i = tid; i < VB * CHUNK; i += 256) {
      int vv = i >> 6, t = i & 63;
      vT[vv * 64 + t] =
          v16[((size_t)(b * TSEQ + c * CHUNK + t)) * DM + (size_t)h * HD + vbase + vv];
    }
    __syncthreads();

    // ---- A = qs @ ki^T  (64x64, K=256), causal mask, store f16 ----
    for (int ti = 0; ti < 2; ++ti) {
      int tileid = wave * 2 + ti;
      int mi = tileid >> 2, ni = tileid & 3;
      v8f a = vzero8();
      for (int s = 0; s < 8; ++s) {
        v16h af = ld_frag(qs + (mi * 16) * 256, 256, lane, s * 32);
        v16h bf = ld_frag(ki + (ni * 16) * 256, 256, lane, s * 32);
        a = wmma16(af, bf, a);
      }
      int n = lane & 15, mr = (lane >> 4) * 8;
      for (int r = 0; r < 8; ++r) {
        int row = mi * 16 + mr + r, col = ni * 16 + n;
        Am[row * 72 + col] = (_Float16)((col <= row) ? a[r] : 0.f);
      }
    }
    __syncthreads();

    // ---- o = A @ v  +  qs @ S (uses previous S16) ----
    {
      int mi = wave >> 1, ni = wave & 1;   // 4 x 2 tiles of 16x16 over 64x32
      v8f a = vzero8();
      for (int s = 0; s < 2; ++s) {
        v16h af = ld_frag(Am + (mi * 16) * 72, 72, lane, s * 32);
        v16h bf = ld_frag(vT + (ni * 16) * 64, 64, lane, s * 32);
        a = wmma16(af, bf, a);
      }
      for (int s = 0; s < 8; ++s) {
        v16h af = ld_frag(qs + (mi * 16) * 256, 256, lane, s * 32);
        v16h bf = ld_frag(S16 + (ni * 16) * 256, 256, lane, s * 32);
        a = wmma16(af, bf, a);
      }
      int n = lane & 15, mr = (lane >> 4) * 8;
      for (int r = 0; r < 8; ++r) {
        int t = mi * 16 + mr + r, vv = ni * 16 + n;
        o[((size_t)(b * TSEQ + c * CHUNK + t)) * DM + (size_t)h * HD + vbase + vv] = a[r];
      }
    }
    __syncthreads();

    // ---- S^T[v][k] = S^T[v][k]*exp(bl[k]) + sum_t vT[v][t]*k_in[t][k] ----
    for (int ti = 0; ti < 4; ++ti) {
      int tileid = wave * 4 + ti;          // 0..31 over 2(v) x 16(k) tiles
      int mi = tileid & 1, ni = tileid >> 1;
      int n = lane & 15, mr = (lane >> 4) * 8;
      v8f a;
      for (int r = 0; r < 8; ++r) {
        int vv = mi * 16 + mr + r, kc = ni * 16 + n;
        a[r] = S[vv * 256 + kc] * ebl[kc];
      }
      for (int s = 0; s < 2; ++s) {
        v16h af = ld_frag(vT + (mi * 16) * 64, 64, lane, s * 32);
        v16h bf = ld_frag(knT + (ni * 16) * 64, 64, lane, s * 32);
        a = wmma16(af, bf, a);
      }
      for (int r = 0; r < 8; ++r) {
        int vv = mi * 16 + mr + r, kc = ni * 16 + n;
        S[vv * 256 + kc] = a[r];
        S16[vv * 256 + kc] = (_Float16)a[r];
      }
    }
    __syncthreads();
  }
}

// ---------------------------------------------------------------------------
// Per-head RMSNorm(o) * silu(g) -> f16. One block per token; loop heads.
// ---------------------------------------------------------------------------
__global__ __launch_bounds__(256)
void glaql_gnorm_gate(const float* __restrict__ o, const float* __restrict__ glin,
                      const float* __restrict__ gw, _Float16* __restrict__ out16) {
  __shared__ float red[256];
  const size_t tok = blockIdx.x;
  const int tid = threadIdx.x;
  for (int h = 0; h < HEADS; ++h) {
    size_t idx = tok * DM + (size_t)h * HD + tid;
    float v = o[idx];
    red[tid] = v * v; __syncthreads();
    for (int st = 128; st > 0; st >>= 1) { if (tid < st) red[tid] += red[tid + st]; __syncthreads(); }
    float var = red[0] / (float)HD;
    __syncthreads();
    float nv = v * rsqrtf(var + 1e-5f) * gw[tid];
    float g = glin[idx];
    out16[idx] = (_Float16)(nv * siluf(g));
  }
}

// ---------------------------------------------------------------------------
// SwiGLU combine: act16 = silu(gate) * up
// ---------------------------------------------------------------------------
__global__ void glaql_act(const float* __restrict__ gate, const float* __restrict__ up,
                          _Float16* __restrict__ act) {
  size_t total = (size_t)NTOK * IM;
  for (size_t i = (size_t)blockIdx.x * blockDim.x + threadIdx.x; i < total;
       i += (size_t)gridDim.x * blockDim.x) {
    act[i] = (_Float16)(siluf(gate[i]) * up[i]);
  }
}

// ---------------------------------------------------------------------------
// Host launch
// ---------------------------------------------------------------------------
extern "C" void kernel_launch(void* const* d_in, const int* in_sizes, int n_in,
                              void* d_out, int out_size, void* d_ws, size_t ws_size,
                              hipStream_t stream) {
  const float* hidden   = (const float*)d_in[0];
  const float* ln1_w    = (const float*)d_in[1];
  const float* ln2_w    = (const float*)d_in[2];
  const float* q_w      = (const float*)d_in[3];
  const float* k_w      = (const float*)d_in[4];
  const float* v_w      = (const float*)d_in[5];
  const float* g_w      = (const float*)d_in[6];
  const float* conv_q   = (const float*)d_in[7];
  const float* conv_k   = (const float*)d_in[8];
  const float* conv_v   = (const float*)d_in[9];
  const float* gk_w1    = (const float*)d_in[10];
  const float* gk_w2    = (const float*)d_in[11];
  const float* gk_b     = (const float*)d_in[12];
  const float* gnorm_w  = (const float*)d_in[13];
  const float* o_w      = (const float*)d_in[14];
  const float* mlp_gate = (const float*)d_in[15];
  const float* mlp_up   = (const float*)d_in[16];
  const float* mlp_down = (const float*)d_in[17];
  float* outp = (float*)d_out;

  char* ws = (char*)d_ws;
  size_t off = 0;
  auto bump = [&](size_t bytes) -> void* {
    void* p = ws + off; off += (bytes + 255) & ~(size_t)255; return p;
  };

  const size_t DD = (size_t)DM * DM, DI = (size_t)DM * IM;
  _Float16* wqT   = (_Float16*)bump(DD * 2);
  _Float16* wkT   = (_Float16*)bump(DD * 2);
  _Float16* wvT   = (_Float16*)bump(DD * 2);
  _Float16* wgT   = (_Float16*)bump(DD * 2);
  _Float16* woT   = (_Float16*)bump(DD * 2);
  _Float16* wgateT = (_Float16*)bump(DI * 2);
  _Float16* wupT   = (_Float16*)bump(DI * 2);
  _Float16* wdownT = (_Float16*)bump(DI * 2);
  _Float16* x16   = (_Float16*)bump((size_t)NTOK * DM * 2);
  float*    lin   = (float*)bump((size_t)NTOK * DM * 4);
  float*    glin  = (float*)bump((size_t)NTOK * DM * 4);
  _Float16* q16   = (_Float16*)bump((size_t)NTOK * DM * 2);
  _Float16* k16   = (_Float16*)bump((size_t)NTOK * DM * 2);
  _Float16* v16b  = (_Float16*)bump((size_t)NTOK * DM * 2);
  float*    gktmp = (float*)bump((size_t)NTOK * 16 * 4);
  float*    gkbuf = (float*)bump((size_t)NTOK * DM * 4);
  float*    obuf  = (float*)bump((size_t)NTOK * DM * 4);
  _Float16* og16  = (_Float16*)bump((size_t)NTOK * DM * 2);
  float*    hbuf  = (float*)bump((size_t)NTOK * DM * 4);
  _Float16* y16   = (_Float16*)bump((size_t)NTOK * DM * 2);
  float*    gateb = (float*)bump((size_t)NTOK * IM * 4);
  float*    upb   = (float*)bump((size_t)NTOK * IM * 4);
  _Float16* act16 = (_Float16*)bump((size_t)NTOK * IM * 2);

  const dim3 blk(256);
  const int GS = 8192;  // grid-stride block count

  // 1. weight conversion (f32 -> transposed f16)
  glaql_convT<<<GS, blk, 0, stream>>>(q_w, wqT, DM, DM);
  glaql_convT<<<GS, blk, 0, stream>>>(k_w, wkT, DM, DM);
  glaql_convT<<<GS, blk, 0, stream>>>(v_w, wvT, DM, DM);
  glaql_convT<<<GS, blk, 0, stream>>>(g_w, wgT, DM, DM);
  glaql_convT<<<GS, blk, 0, stream>>>(o_w, woT, DM, DM);
  glaql_convT<<<GS, blk, 0, stream>>>(mlp_gate, wgateT, DM, IM);
  glaql_convT<<<GS, blk, 0, stream>>>(mlp_up,   wupT,   DM, IM);
  glaql_convT<<<GS, blk, 0, stream>>>(mlp_down, wdownT, IM, DM);

  // 2. ln1
  glaql_rmsnorm16<<<NTOK, blk, 0, stream>>>(hidden, ln1_w, x16, 1e-6f);

  // 3-5. q/k/v projections + conv+silu (lin buffer reused)
  dim3 gD(DM / GBN, NTOK / GBM);
  glaql_gemm<0><<<gD, blk, 0, stream>>>(x16, wqT, lin, nullptr, NTOK, DM, DM);
  glaql_conv_silu<<<GS, blk, 0, stream>>>(lin, conv_q, q16);
  glaql_gemm<0><<<gD, blk, 0, stream>>>(x16, wkT, lin, nullptr, NTOK, DM, DM);
  glaql_conv_silu<<<GS, blk, 0, stream>>>(lin, conv_k, k16);
  glaql_gemm<0><<<gD, blk, 0, stream>>>(x16, wvT, lin, nullptr, NTOK, DM, DM);
  glaql_conv_silu<<<GS, blk, 0, stream>>>(lin, conv_v, v16b);

  // 6. g projection
  glaql_gemm<0><<<gD, blk, 0, stream>>>(x16, wgT, glin, nullptr, NTOK, DM, DM);

  // 7. gk path
  glaql_gk1<<<NTOK, blk, 0, stream>>>(x16, gk_w1, gktmp);
  glaql_gk2<<<GS, blk, 0, stream>>>(gktmp, gk_w2, gk_b, gkbuf);

  // 8. chunked GLA
  const size_t gla_smem =
      (size_t)(VB * 256 + 256) * 4 +
      (size_t)(VB * 256 + 64 * 256 + 64 * 256 + 256 * 64 + VB * 64 + 64 * 72) * 2;
  (void)hipFuncSetAttribute((const void*)glaql_gla,
                            hipFuncAttributeMaxDynamicSharedMemorySize, (int)gla_smem);
  glaql_gla<<<dim3(BB * HEADS, HD / VB), blk, gla_smem, stream>>>(q16, k16, v16b, gkbuf, obuf);

  // 9. head-RMSNorm * silu(g)
  glaql_gnorm_gate<<<NTOK, blk, 0, stream>>>(obuf, glin, gnorm_w, og16);

  // 10. o-projection + residual -> h
  glaql_gemm<1><<<gD, blk, 0, stream>>>(og16, woT, hbuf, hidden, NTOK, DM, DM);

  // 11. ln2
  glaql_rmsnorm16<<<NTOK, blk, 0, stream>>>(hbuf, ln2_w, y16, 1e-6f);

  // 12. MLP
  dim3 gI(IM / GBN, NTOK / GBM);
  glaql_gemm<0><<<gI, blk, 0, stream>>>(y16, wgateT, gateb, nullptr, NTOK, IM, DM);
  glaql_gemm<0><<<gI, blk, 0, stream>>>(y16, wupT,   upb,   nullptr, NTOK, IM, DM);
  glaql_act<<<GS, blk, 0, stream>>>(gateb, upb, act16);

  // 13. down-projection + residual -> out
  glaql_gemm<1><<<gD, blk, 0, stream>>>(act16, wdownT, outp, hbuf, NTOK, DM, IM);
}
